// POTBarycentricPooling_39900246180413
// MI455X (gfx1250) — compile-verified, matrix operand order checked by reading, not run on hardware
//
#include <hip/hip_runtime.h>
#include <math.h>

typedef __attribute__((ext_vector_type(16))) _Float16 v16h;
typedef __attribute__((ext_vector_type(8)))  float    v8f;

#define EPS_REG   0.2f
#define MAX_ITER  100
#define STAB_EPS  1e-8f
#define DSZ 4      // DIST_SIZE
#define HD  128    // HIDDEN
#define KK  16     // CODEBOOK_SIZE
#define BB  64     // BATCH

#define SK_THREADS 512          // Sinkhorn block: 16 waves -> 4 waves/SIMD32
#define SK_NW      (SK_THREADS / 32)

__device__ __forceinline__ float sq4(float4 a) {
  return a.x*a.x + a.y*a.y + a.z*a.z + a.w*a.w;
}

__device__ __forceinline__ void load_row16(const float* __restrict__ p, float t[KK]) {
  const float4* r4 = (const float4*)p;
  float4 r0 = r4[0], r1 = r4[1], r2 = r4[2], r3 = r4[3];
  t[0]=r0.x;  t[1]=r0.y;  t[2]=r0.z;  t[3]=r0.w;
  t[4]=r1.x;  t[5]=r1.y;  t[6]=r1.z;  t[7]=r1.w;
  t[8]=r2.x;  t[9]=r2.y;  t[10]=r2.z; t[11]=r2.w;
  t[12]=r3.x; t[13]=r3.y; t[14]=r3.z; t[15]=r3.w;
}

// ---------------- tiny setup kernels ----------------
__global__ void k_zero_counts(int* counts) {
  if (threadIdx.x < BB) counts[threadIdx.x] = 0;
}

__global__ void k_hist(const int* __restrict__ bidx, int* counts, int n) {
  int i = blockIdx.x * blockDim.x + threadIdx.x;
  if (i < n) atomicAdd(&counts[bidx[i]], 1);
}

__global__ void k_scan(const int* __restrict__ counts, int* starts) {
  if (threadIdx.x == 0) {
    int acc = 0;
    for (int b = 0; b < BB; ++b) { starts[b] = acc; acc += counts[b]; }
    starts[BB] = acc;
  }
}

// ---------------- stage 1: logK = -max(||x||^2 + ||c||^2 - 2 x.c, 0)/eps ----
// one wave = one 16-row tile; 4x v_wmma_f32_16x16x32_f16 over K=128
__global__ __launch_bounds__(256) void k_cost(const float* __restrict__ X,
                                              const float* __restrict__ C,
                                              float* __restrict__ logK, int ns) {
  __shared__ _Float16 cb[KK * HD];
  __shared__ float cn[KK];
  __shared__ float xn[8][16];

  for (int idx = threadIdx.x; idx < KK * HD; idx += 256)
    cb[idx] = (_Float16)C[idx];
  if (threadIdx.x < KK) {
    float s = 0.f;
    for (int d = 0; d < HD; ++d) { float c = C[threadIdx.x * HD + d]; s += c * c; }
    cn[threadIdx.x] = s;
  }
  __syncthreads();

  const int wave = threadIdx.x >> 5;
  const int lane = threadIdx.x & 31;
  const int nsel = lane & 15;  // column (codeword) / row-within-tile selector
  const int hh   = lane >> 4;  // half selector
  const long long tile = (long long)blockIdx.x * 8 + wave;
  const long long rowBase = tile * 16;
  if (rowBase >= ns) return;

  // B operand: codebook^T 32x16 f16 slices, packed once per wave from LDS
  v16h bt[4];
#pragma unroll
  for (int kb = 0; kb < 4; ++kb) {
#pragma unroll
    for (int j = 0; j < 16; ++j) {
      int k = kb * 32 + ((j < 8) ? (hh * 8 + j) : (16 + hh * 8 + (j - 8)));
      bt[kb][j] = cb[nsel * HD + k];
    }
  }

  long long r = rowBase + nsel;
  if (r >= ns) r = ns - 1;  // clamp; stores are predicated below
  const float4* Xr = (const float4*)(X + r * HD);

  v8f acc = {};
  float sq = 0.f;
#pragma unroll
  for (int kb = 0; kb < 4; ++kb) {
    const int k0 = kb * 32 + hh * 8;
    float4 p0 = Xr[k0 / 4], p1 = Xr[k0 / 4 + 1];
    float4 q0 = Xr[k0 / 4 + 4], q1 = Xr[k0 / 4 + 5];
    sq += sq4(p0) + sq4(p1) + sq4(q0) + sq4(q1);
    v16h a;
    a[0]=(_Float16)p0.x;  a[1]=(_Float16)p0.y;  a[2]=(_Float16)p0.z;  a[3]=(_Float16)p0.w;
    a[4]=(_Float16)p1.x;  a[5]=(_Float16)p1.y;  a[6]=(_Float16)p1.z;  a[7]=(_Float16)p1.w;
    a[8]=(_Float16)q0.x;  a[9]=(_Float16)q0.y;  a[10]=(_Float16)q0.z; a[11]=(_Float16)q0.w;
    a[12]=(_Float16)q1.x; a[13]=(_Float16)q1.y; a[14]=(_Float16)q1.z; a[15]=(_Float16)q1.w;
    acc = __builtin_amdgcn_wmma_f32_16x16x32_f16(false, a, false, bt[kb],
                                                 (short)0, acc, false, false);
  }

  // ||x||^2 per row: lane l and lane l^16 cover disjoint K halves of row l%16
  float x2 = sq + __shfl_xor(sq, 16, 32);
  if (lane < 16) xn[wave][lane] = x2;   // wave-internal LDS, in-order DS pipe

  const float cnv = cn[nsel];
#pragma unroll
  for (int v = 0; v < 8; ++v) {
    int mloc = v + 8 * hh;
    long long rr = rowBase + mloc;
    if (rr < ns) {
      float dot = acc[v];
      float Mc = xn[wave][mloc] + cnv - 2.0f * dot;
      Mc = fmaxf(Mc, 0.0f);
      logK[rr * KK + nsel] = -Mc * (1.0f / EPS_REG);
    }
  }
}

// ---------------- stage 2+3: persistent Sinkhorn per graph ----------------
__global__ __launch_bounds__(SK_THREADS) void k_sinkhorn(
    const float* __restrict__ logK, const int* __restrict__ starts,
    float* __restrict__ fbuf, float* __restrict__ out) {
  const int b = blockIdx.x;
  const int tid = threadIdx.x;
  const int lane = tid & 31, wave = tid >> 5;
  __shared__ float g_sh[KK];
  __shared__ float rm[KK][SK_NW], rs[KK][SK_NW];
  __shared__ float tot_sh;

  const int n0 = starts[b], n1 = starts[b + 1];
  const int cnt = n1 - n0;
  if (cnt == 0) {  // empty graph -> zero row (block-uniform early exit)
    if (tid < KK) out[b * KK + tid] = 0.f;
    return;
  }
  const int s0 = n0 * DSZ, s1 = n1 * DSZ;
  const float log_a = -__logf((float)cnt * (float)DSZ);
  const float log_b = -__logf((float)KK);
  if (tid < KK) g_sh[tid] = 0.f;
  __syncthreads();

  for (int it = 0; it < MAX_ITER; ++it) {
    float gr[KK];
#pragma unroll
    for (int k = 0; k < KK; ++k) gr[k] = g_sh[k];
    float mk[KK], sk[KK];
#pragma unroll
    for (int k = 0; k < KK; ++k) { mk[k] = -__builtin_inff(); sk[k] = 0.f; }

    // fused pass: f_i from current g, fold into online LSE for new g
    for (int i = s0 + tid; i < s1; i += SK_THREADS) {
      float t[KK];
      load_row16(logK + (long long)i * KK, t);
      float mm = -__builtin_inff();
#pragma unroll
      for (int k = 0; k < KK; ++k) mm = fmaxf(mm, t[k] + gr[k]);
      float se = 0.f;
#pragma unroll
      for (int k = 0; k < KK; ++k) se += __expf(t[k] + gr[k] - mm);
      float fi = log_a - (mm + __logf(se));
      if (it == MAX_ITER - 1) fbuf[i] = fi;  // read back by the SAME thread later
#pragma unroll
      for (int k = 0; k < KK; ++k) {
        float v = t[k] + fi;
        if (v > mk[k]) { sk[k] = sk[k] * __expf(mk[k] - v) + 1.f; mk[k] = v; }
        else           { sk[k] += __expf(v - mk[k]); }
      }
    }

    // wave32 butterfly merge of (m, s) pairs
#pragma unroll
    for (int off = 16; off >= 1; off >>= 1) {
#pragma unroll
      for (int k = 0; k < KK; ++k) {
        float m2 = __shfl_xor(mk[k], off, 32);
        float s2 = __shfl_xor(sk[k], off, 32);
        if (s2 > 0.f) {
          if (sk[k] <= 0.f)      { mk[k] = m2; sk[k] = s2; }
          else if (m2 > mk[k])   { sk[k] = sk[k] * __expf(mk[k] - m2) + s2; mk[k] = m2; }
          else                   { sk[k] += s2 * __expf(m2 - mk[k]); }
        }
      }
    }
    if (lane == 0) {
#pragma unroll
      for (int k = 0; k < KK; ++k) { rm[k][wave] = mk[k]; rs[k][wave] = sk[k]; }
    }
    __syncthreads();
    if (tid < KK) {
      float m = -__builtin_inff(), s = 0.f;
#pragma unroll
      for (int w = 0; w < SK_NW; ++w) {
        float m2 = rm[tid][w], s2 = rs[tid][w];
        if (s2 > 0.f) {
          if (s <= 0.f)    { m = m2; s = s2; }
          else if (m2 > m) { s = s * __expf(m - m2) + s2; m = m2; }
          else             { s += s2 * __expf(m2 - m); }
        }
      }
      float mfin = (m > -1e37f && m < 1e37f) ? m : 0.f;  // where(isfinite(m), m, 0)
      g_sh[tid] = log_b - (mfin + __logf(s + 1e-30f));
    }
    __syncthreads();
  }

  // transport-plan column marginal: sum_i exp(f_i + g_k + logK[i,k])
  float gr[KK];
#pragma unroll
  for (int k = 0; k < KK; ++k) gr[k] = g_sh[k];
  float ak[KK];
#pragma unroll
  for (int k = 0; k < KK; ++k) ak[k] = 0.f;
  for (int i = s0 + tid; i < s1; i += SK_THREADS) {
    float t[KK];
    load_row16(logK + (long long)i * KK, t);
    float fi = fbuf[i];
#pragma unroll
    for (int k = 0; k < KK; ++k) ak[k] += __expf(fi + gr[k] + t[k]);
  }
#pragma unroll
  for (int off = 16; off >= 1; off >>= 1) {
#pragma unroll
    for (int k = 0; k < KK; ++k) ak[k] += __shfl_xor(ak[k], off, 32);
  }
  if (lane == 0) {
#pragma unroll
    for (int k = 0; k < KK; ++k) rm[k][wave] = ak[k];
  }
  __syncthreads();
  if (tid < KK) {
    float m = 0.f;
#pragma unroll
    for (int w = 0; w < SK_NW; ++w) m += rm[tid][w];
    rs[tid][0] = m;
  }
  __syncthreads();
  if (tid == 0) {
    float t = 0.f;
#pragma unroll
    for (int k = 0; k < KK; ++k) t += rs[k][0];
    tot_sh = fmaxf(t, STAB_EPS);
  }
  __syncthreads();
  if (tid < KK) out[b * KK + tid] = rs[tid][0] / tot_sh;
}

// ---------------- host launcher ----------------
extern "C" void kernel_launch(void* const* d_in, const int* in_sizes, int n_in,
                              void* d_out, int out_size, void* d_ws, size_t ws_size,
                              hipStream_t stream) {
  const float* X   = (const float*)d_in[0];  // [N, 4, 128] f32
  const float* C   = (const float*)d_in[1];  // [16, 128]  f32
  const int* bidx  = (const int*)d_in[2];    // [N] int32 (sorted)
  const int n  = in_sizes[2];
  const int ns = n * DSZ;

  // workspace layout
  float* logK  = (float*)d_ws;                       // ns*16 f32 (25.6 MB)
  float* fbuf  = logK + (size_t)ns * KK;             // ns f32
  int*   counts = (int*)(fbuf + ns);                 // 64 i32
  int*   starts = counts + BB;                       // 65 i32
  float* out = (float*)d_out;                        // [64,16] f32

  k_zero_counts<<<1, 64, 0, stream>>>(counts);
  k_hist<<<(n + 255) / 256, 256, 0, stream>>>(bidx, counts, n);
  k_scan<<<1, 64, 0, stream>>>(counts, starts);

  const int tiles  = (ns + 15) / 16;
  const int blocks = (tiles + 7) / 8;   // 8 waves/block, 1 tile/wave
  k_cost<<<blocks, 256, 0, stream>>>(X, C, logK, ns);

  k_sinkhorn<<<BB, SK_THREADS, 0, stream>>>(logK, starts, fbuf, out);
}
